// MultiAttentionLayer_58085137711796
// MI455X (gfx1250) — matrix operand order repaired and checked by reading, not compiled, not run on hardware
//
#include <hip/hip_runtime.h>
#include <hip/hip_bf16.h>

// ---------------------------------------------------------------------------
// Multi-head graph attention, MI455X (gfx1250, wave32, WMMA).
//   H=4 heads, N=6144 tokens, IN=256, OUT=64.
// Pipeline:
//   1) proj_kernel: Q = hWQ+b, K = hWK+b (f16 row-major [H][N][64]),
//      V = hWV+b stored transposed Vt [H][64][N] f16  -- all via
//      v_wmma_f32_16x16x32_f16.
//   2) attn_kernel: per wave = one (head, 32-query-row block = two 16-row
//      WMMA tiles sharing K/V B-tiles). Streams adj once with NT hints
//      (protects L2-resident K/V from the 604MB stream), computes S tiles
//      with WMMA, exp without max-subtraction (adj in [0,1) bounds logits),
//      per-wave LDS C->A relayout, P@V with WMMA, row-sum normalize.
// ---------------------------------------------------------------------------

#define H 4
#define NTOK 6144
#define IND 256
#define OUTD 64

typedef __attribute__((ext_vector_type(16))) _Float16 v16h;
typedef __attribute__((ext_vector_type(8)))  float    v8f;

// 16-bit A-matrix 16x32 (MxK): VGPR p (pair 2p,2p+1 of v16h), lane half:
//   p<4 : K = 2p + 8*half ;  p>=4: K = 16 + 2(p-4) + 8*half
__device__ __forceinline__ int a_pair_k(int p, int half) {
    return (p < 4) ? (2 * p + 8 * half) : (16 + 2 * (p - 4) + 8 * half);
}
// 16-bit B-matrix 32x16 (KxN): same K order as A (pairs 2p,2p+1):
//   p<4 : K = 2p + 8*half ;  p>=4: K = 16 + 2(p-4) + 8*half; N = lane & 15.
__device__ __forceinline__ int b_pair_k(int p, int half) {
    return (p < 4) ? (2 * p + 8 * half) : (16 + 2 * (p - 4) + 8 * half);
}

// ---------------------------------------------------------------------------
// Projection kernel: tasks = H * 3(Q/K/V) * (N/16) row-tiles, one per wave.
// ---------------------------------------------------------------------------
__launch_bounds__(256)
__global__ void proj_kernel(const float* __restrict__ h,
                            const float* __restrict__ WQ, const float* __restrict__ bQ,
                            const float* __restrict__ WK, const float* __restrict__ bK,
                            const float* __restrict__ WV, const float* __restrict__ bV,
                            _Float16* __restrict__ Qh, _Float16* __restrict__ Kh,
                            _Float16* __restrict__ Vt)
{
    const int lane = threadIdx.x & 31;
    const int wv   = threadIdx.x >> 5;
    const int task = blockIdx.x * 8 + wv;          // uniform per wave
    const int qt    = task % (NTOK / 16);
    const int rem   = task / (NTOK / 16);
    const int which = rem % 3;
    const int hd    = rem / 3;
    if (hd >= H) return;                           // whole-wave uniform exit

    const int m = lane & 15, half = lane >> 4;
    const int qbase = qt * 16;

    const float* W    = (which == 0) ? WQ : (which == 1) ? WK : WV;
    const float* bias = (which == 0) ? bQ : (which == 1) ? bK : bV;
    W    += (size_t)hd * IND * OUTD;
    bias += hd * OUTD;

    v8f c[4];
#pragma unroll
    for (int j = 0; j < 4; ++j) c[j] = (v8f){0.f,0.f,0.f,0.f,0.f,0.f,0.f,0.f};

    for (int kc = 0; kc < IND; kc += 32) {
        v16h a;
        const float* hrow = h + (size_t)(qbase + m) * IND + kc;
#pragma unroll
        for (int p = 0; p < 8; ++p) {
            const int k0 = a_pair_k(p, half);
            a[2 * p]     = (_Float16)hrow[k0];
            a[2 * p + 1] = (_Float16)hrow[k0 + 1];
        }
#pragma unroll
        for (int j = 0; j < 4; ++j) {
            const int n = j * 16 + m;
            v16h b;
#pragma unroll
            for (int p = 0; p < 8; ++p) {
                const int k0 = kc + b_pair_k(p, half);
                b[2 * p]     = (_Float16)W[(size_t)k0 * OUTD + n];
                b[2 * p + 1] = (_Float16)W[(size_t)(k0 + 1) * OUTD + n];
            }
            c[j] = __builtin_amdgcn_wmma_f32_16x16x32_f16(
                false, a, false, b, (short)0, c[j], false, false);
        }
    }

    // bias + store (C/D layout: row = r + 8*half, col = j*16 + m)
#pragma unroll
    for (int j = 0; j < 4; ++j) {
        const int n = j * 16 + m;
        const float bv = bias[n];
#pragma unroll
        for (int r = 0; r < 8; ++r) {
            const int row = qbase + r + 8 * half;
            const float val = c[j][r] + bv;
            if (which == 2) {
                Vt[((size_t)hd * OUTD + n) * NTOK + row] = (_Float16)val;
            } else {
                _Float16* dst = (which == 0) ? Qh : Kh;
                dst[((size_t)hd * NTOK + row) * OUTD + n] = (_Float16)val;
            }
        }
    }
}

// ---------------------------------------------------------------------------
// Attention kernel: one wave = one (head, 32-row query block), two 16-row
// WMMA A-tiles share every K/V B-tile (halves L2 traffic, 16 WMMAs per
// 32-key step). 8 waves / block, 2KB LDS per wave.
// ---------------------------------------------------------------------------
__launch_bounds__(256)
__global__ void attn_kernel(const float* __restrict__ adj,
                            const _Float16* __restrict__ Qh,
                            const _Float16* __restrict__ Kh,
                            const _Float16* __restrict__ Vt,
                            float* __restrict__ X)
{
    __shared__ _Float16 lds[8 * 2 * 16 * 32];      // 16 KB: 2 P patches / wave
    const int lane = threadIdx.x & 31;
    const int wv   = threadIdx.x >> 5;
    const int task = blockIdx.x * 8 + wv;          // 0 .. H*192-1
    const int qt = task % (NTOK / 32);
    const int hd = task / (NTOK / 32);
    if (hd >= H) return;

    const int m = lane & 15, half = lane >> 4;
    const int qbase = qt * 32;
    _Float16* pl = lds + wv * (2 * 16 * 32);
    const float scale = 0.0625f;                    // 1/sqrt(256)

    // Two Q A-tiles (rows qbase+16t), both 32-wide K chunks of the 64 dim
    v16h aq[2][2];
#pragma unroll
    for (int t = 0; t < 2; ++t) {
        const _Float16* qrow =
            Qh + ((size_t)hd * NTOK + qbase + 16 * t + m) * OUTD;
#pragma unroll
        for (int cH = 0; cH < 2; ++cH)
#pragma unroll
            for (int p = 0; p < 8; ++p) {
                const int k0 = cH * 32 + a_pair_k(p, half);
                aq[t][cH][2 * p]     = qrow[k0];
                aq[t][cH][2 * p + 1] = qrow[k0 + 1];
            }
    }

    v8f acc[2][4];
#pragma unroll
    for (int t = 0; t < 2; ++t)
#pragma unroll
        for (int j = 0; j < 4; ++j)
            acc[t][j] = (v8f){0.f,0.f,0.f,0.f,0.f,0.f,0.f,0.f};
    float rs[2][8];
#pragma unroll
    for (int t = 0; t < 2; ++t)
#pragma unroll
        for (int r = 0; r < 8; ++r) rs[t][r] = 0.f;

    const size_t adj_base = ((size_t)hd * NTOK + qbase) * (size_t)NTOK;

    for (int kt = 0; kt < NTOK; kt += 32) {
        // Prefetch next adj tile (global_prefetch_b8); 32 lanes = 32 rows
        if (kt + 32 < NTOK)
            __builtin_prefetch(adj + adj_base + (size_t)lane * NTOK + kt + 32, 0, 1);

#pragma unroll
        for (int s = 0; s < 2; ++s) {
            const int kb = kt + s * 16;
            // Shared K B-tiles for both query A-tiles
            v16h bk[2];
#pragma unroll
            for (int cH = 0; cH < 2; ++cH) {
                const _Float16* krow =
                    Kh + ((size_t)hd * NTOK + kb + m) * OUTD + cH * 32;
#pragma unroll
                for (int p = 0; p < 8; ++p) {
                    const int f0 = b_pair_k(p, half);
                    bk[cH][2 * p]     = krow[f0];
                    bk[cH][2 * p + 1] = krow[f0 + 1];
                }
            }
            v8f c[2];
#pragma unroll
            for (int t = 0; t < 2; ++t) {
                c[t] = (v8f){0.f,0.f,0.f,0.f,0.f,0.f,0.f,0.f};
                c[t] = __builtin_amdgcn_wmma_f32_16x16x32_f16(
                    false, aq[t][0], false, bk[0], (short)0, c[t], false, false);
                c[t] = __builtin_amdgcn_wmma_f32_16x16x32_f16(
                    false, aq[t][1], false, bk[1], (short)0, c[t], false, false);
            }
            // adj mask (NT stream: don't thrash L2 holding K/V), exp, P->LDS
#pragma unroll
            for (int t = 0; t < 2; ++t)
#pragma unroll
                for (int r = 0; r < 8; ++r) {
                    const size_t idx = adj_base +
                        (size_t)(16 * t + r + 8 * half) * NTOK + kb + m;
                    const float av = __builtin_nontemporal_load(adj + idx);
                    const float pv = __expf(c[t][r] * scale * av);
                    rs[t][r] += pv;
                    pl[t * 512 + (r + 8 * half) * 32 + s * 16 + m] = (_Float16)pv;
                }
        }

        // Re-read P in A layout (DS ops in-order within a wave)
        v16h pa[2];
#pragma unroll
        for (int t = 0; t < 2; ++t)
#pragma unroll
            for (int p = 0; p < 8; ++p) {
                const int k0 = a_pair_k(p, half);
                pa[t][2 * p]     = pl[t * 512 + m * 32 + k0];
                pa[t][2 * p + 1] = pl[t * 512 + m * 32 + k0 + 1];
            }
        // P(16x32) @ V(32x64): shared V B-tile feeds both P A-tiles
#pragma unroll
        for (int j = 0; j < 4; ++j) {
            v16h bv;
            const _Float16* vrow =
                Vt + ((size_t)hd * OUTD + j * 16 + m) * NTOK + kt;
#pragma unroll
            for (int p = 0; p < 8; ++p) {
                const int k0 = b_pair_k(p, half);
                bv[2 * p]     = vrow[k0];
                bv[2 * p + 1] = vrow[k0 + 1];
            }
#pragma unroll
            for (int t = 0; t < 2; ++t)
                acc[t][j] = __builtin_amdgcn_wmma_f32_16x16x32_f16(
                    false, pa[t], false, bv, (short)0, acc[t][j], false, false);
        }
    }

    // Row-sum butterfly across the 16 lanes holding one row's columns,
    // then normalized NT store of X [H][N][64] f32 (written once).
#pragma unroll
    for (int t = 0; t < 2; ++t) {
#pragma unroll
        for (int r = 0; r < 8; ++r) {
            float v = rs[t][r];
#pragma unroll
            for (int off = 8; off >= 1; off >>= 1)
                v += __shfl_xor(v, off, 16);
            rs[t][r] = 1.0f / v;
        }
#pragma unroll
        for (int j = 0; j < 4; ++j)
#pragma unroll
            for (int r = 0; r < 8; ++r) {
                const int row = qbase + 16 * t + r + 8 * half;
                __builtin_nontemporal_store(
                    acc[t][j][r] * rs[t][r],
                    X + ((size_t)hd * NTOK + row) * OUTD + j * 16 + m);
            }
    }
}

extern "C" void kernel_launch(void* const* d_in, const int* in_sizes, int n_in,
                              void* d_out, int out_size, void* d_ws, size_t ws_size,
                              hipStream_t stream) {
    const float* adj = (const float*)d_in[0];
    const float* h   = (const float*)d_in[1];
    const float* WQ  = (const float*)d_in[2];
    const float* bQ  = (const float*)d_in[3];
    const float* WK  = (const float*)d_in[4];
    const float* bK  = (const float*)d_in[5];
    const float* WV  = (const float*)d_in[6];
    const float* bV  = (const float*)d_in[7];
    float* X = (float*)d_out;

    // Workspace: Qh | Kh (f16 [H][N][64]) | Vt (f16 [H][64][N]) = 9 MB total
    const size_t elems = (size_t)H * NTOK * OUTD;
    _Float16* Qh = (_Float16*)d_ws;
    _Float16* Kh = Qh + elems;
    _Float16* Vt = Kh + elems;

    // H*3*(N/16) = 4608 wave-tasks, 8 waves per 256-thread block
    proj_kernel<<<576, 256, 0, stream>>>(h, WQ, bQ, WK, bK, WV, bV, Qh, Kh, Vt);
    // H*(N/32) = 768 wave-tasks
    attn_kernel<<<96, 256, 0, stream>>>(adj, Qh, Kh, Vt, X);
    (void)in_sizes; (void)n_in; (void)out_size; (void)ws_size;
}
